// GraphBondingNetwork_82592221102563
// MI455X (gfx1250) — matrix operand order repaired and checked by reading, not compiled, not run on hardware
//
#include <hip/hip_runtime.h>
#include <hip/hip_bf16.h>
#include <math.h>

typedef __attribute__((ext_vector_type(16))) __bf16 v16bf;
typedef __attribute__((ext_vector_type(8)))  __bf16 bf16x8;
typedef __attribute__((ext_vector_type(8)))  float  v8f;

#define NB 4
#define NN 256
#define ND 128
#define NPAIR 32640               // N*(N-1)/2
#define TOTROWS (NB * NPAIR)      // 130560
#define CHUNK 16384               // pair rows per tail chunk (multiple of 16)
#define NITERS 3

// ---------------------------------------------------------------- helpers
static __device__ __forceinline__ __bf16 f2bf(float f) {
  unsigned u = __builtin_bit_cast(unsigned, f);
  unsigned r = (u + 0x7FFFu + ((u >> 16) & 1u)) >> 16;   // round-to-nearest-even
  unsigned short s = (unsigned short)r;
  return __builtin_bit_cast(__bf16, s);
}
static __device__ __forceinline__ float geluf(float x) {
  return 0.5f * x * (1.0f + erff(x * 0.70710678118654752f));
}
static __device__ __forceinline__ float sigmf(float x) {
  return 1.0f / (1.0f + expf(-x));
}

// ---------------------------------------------------------------- WMMA GEMM
// C[m,n] = sum_k A[m,k]*BT[n,k] (+ bias[n]*rowscale[m]) (opt gelu) -> f32 and/or bf16
// Preconditions (guaranteed by caller): M%16==0, N%(16*NT)==0, Kpad%32==0,
// A is bf16 row-major [M][lda] (16B-aligned rows), BT is bf16 [N][ldb] (32B-aligned rows).
// One wave computes a 16 x (16*NT) strip: 1 A fragment -> NT WMMAs per K-step.
// Fragment layouts (wave32); A/B K order is this kernel's contiguous choice, not the ISA table order:
//   A (16x32): lane L -> row L&15, half=L>>4; elem e -> K=16*half+e
//   B (32x16): lane L -> col L&15; elem e -> K=16*half+e   (contiguous in BT row)
//   C (16x16 f32): lane L -> col L&15; vgpr r -> row 8*half+r
template <int NT>
__global__ __launch_bounds__(256)
void wmma_gemm_bf16(const __bf16* __restrict__ A, int lda,
                    const __bf16* __restrict__ BT, int ldb,
                    const float* __restrict__ bias,
                    const float* __restrict__ rowscale,
                    float* __restrict__ Cf, __bf16* __restrict__ Cbf, int ldc,
                    int M, int N, int Kpad, int act_gelu) {
  const int Ngrp  = N / (16 * NT);
  const int tiles = (M >> 4) * Ngrp;
  const int wid   = threadIdx.x >> 5;
  const int tile  = blockIdx.x * 8 + wid;
  if (tile >= tiles) return;
  const int lane = threadIdx.x & 31;
  const int half = lane >> 4;
  const int l16  = lane & 15;
  const int mt = tile / Ngrp, ng = tile % Ngrp;
  const int m  = mt * 16 + l16;                 // A row this lane streams
  const int n0 = ng * (16 * NT) + l16;          // first column this lane owns

  const __bf16* Arow = A + (size_t)m * lda;
  v8f acc[NT];
#pragma unroll
  for (int t = 0; t < NT; ++t) acc[t] = (v8f){};

  for (int kb = 0; kb < Kpad; kb += 32) {
    // A fragment: two contiguous 16B loads, merged
    bf16x8 a0 = *(const bf16x8*)(Arow + kb + half * 16);
    bf16x8 a1 = *(const bf16x8*)(Arow + kb + half * 16 + 8);
    v16bf a = __builtin_shufflevector(a0, a1, 0, 1, 2, 3, 4, 5, 6, 7,
                                      8, 9, 10, 11, 12, 13, 14, 15);
#pragma unroll
    for (int t = 0; t < NT; ++t) {
      // B fragment: one contiguous 32B load from pre-transposed bf16 weights
      v16bf b = *(const v16bf*)(BT + (size_t)(n0 + t * 16) * ldb + kb + half * 16);
      acc[t] = __builtin_amdgcn_wmma_f32_16x16x32_bf16(false, a, false, b,
                                                       (short)0, acc[t], false, false);
    }
  }
#pragma unroll
  for (int t = 0; t < NT; ++t) {
    const int n = n0 + t * 16;
    const float bn = bias ? bias[n] : 0.0f;
#pragma unroll
    for (int r = 0; r < 8; ++r) {
      int mm = mt * 16 + half * 8 + r;
      float v = acc[t][r];
      if (bias) v += bn * (rowscale ? rowscale[mm] : 1.0f);
      if (act_gelu) v = geluf(v);
      if (Cf)  Cf[(size_t)mm * ldc + n]  = v;
      if (Cbf) Cbf[(size_t)mm * ldc + n] = f2bf(v);
    }
  }
}

static void gemm(hipStream_t s, int M, int N, int Kpad,
                 const __bf16* A, int lda, const __bf16* BT, int ldb,
                 const float* bias, const float* rowscale,
                 float* Cf, __bf16* Cbf, int ldc, bool gelu_act) {
  if (N % 64 == 0) {
    int tiles = (M / 16) * (N / 64);
    wmma_gemm_bf16<4><<<dim3((tiles + 7) / 8), dim3(256), 0, s>>>(
        A, lda, BT, ldb, bias, rowscale, Cf, Cbf, ldc, M, N, Kpad, (int)gelu_act);
  } else {
    int tiles = (M / 16) * (N / 32);
    wmma_gemm_bf16<2><<<dim3((tiles + 7) / 8), dim3(256), 0, s>>>(
        A, lda, BT, ldb, bias, rowscale, Cf, Cbf, ldc, M, N, Kpad, (int)gelu_act);
  }
}

// ---------------------------------------------------------------- weight prep
// BT[n*Kpad + k] = (k<K) ? (trans ? W[n*ldw+k] : W[k*ldw+n]) : 0   (bf16)
__global__ void makeBT_kernel(const float* __restrict__ W, int ldw, int trans,
                              int K, int Kpad, int N, __bf16* __restrict__ BT) {
  int idx = blockIdx.x * blockDim.x + threadIdx.x;
  if (idx >= N * Kpad) return;
  int n = idx / Kpad, k = idx % Kpad;
  float v = 0.0f;
  if (k < K) v = trans ? W[(size_t)n * ldw + k] : W[(size_t)k * ldw + n];
  BT[idx] = f2bf(v);
}

// h (f32 state) and h_bf (bf16 GEMM operand) from flat
__global__ void inith_kernel(const float* __restrict__ flat,
                             float* __restrict__ h, __bf16* __restrict__ h_bf) {
  int idx = blockIdx.x * blockDim.x + threadIdx.x;
  if (idx >= NB * NN * ND) return;
  float v = flat[idx];
  h[idx] = v; h_bf[idx] = f2bf(v);
}

// bf16 A operands for the bilinear GEMMs: bond k=0 slice (K=32), charge padded 16->32
__global__ void initvc_kernel(const float* __restrict__ bond, const float* __restrict__ chg,
                              __bf16* __restrict__ abond, __bf16* __restrict__ achg) {
  int idx = blockIdx.x * blockDim.x + threadIdx.x;   // over 1024*32
  if (idx >= NB * NN * 32) return;
  int m = idx >> 5, k = idx & 31;
  abond[idx] = f2bf(bond[(size_t)m * 128 + k]);      // bond_vectors[b,n,0,k]
  achg[idx]  = (k < 16) ? f2bf(chg[(size_t)m * 16 + k])
                        : __builtin_bit_cast(__bf16, (unsigned short)0);
}

// ---------------------------------------------------------------- E0: triu indices
__global__ void pairidx_kernel(int* __restrict__ iu, int* __restrict__ ju) {
  int p = blockIdx.x * blockDim.x + threadIdx.x;
  if (p >= NPAIR) return;
  double disc = (double)(2 * NN - 1) * (2 * NN - 1) - 8.0 * (double)p;
  int i = (int)(((double)(2 * NN - 1) - sqrt(disc)) * 0.5);
  if (i < 0) i = 0;
  if (i > NN - 2) i = NN - 2;
  while (i > 0 && (i * (2 * NN - 1 - i)) / 2 > p) --i;
  while (i < NN - 2 && ((i + 1) * (2 * NN - 1 - (i + 1))) / 2 <= p) ++i;
  int j = p - (i * (2 * NN - 1 - i)) / 2 + i + 1;
  iu[p] = i; ju[p] = j;
}

// ---------------------------------------------------------------- E1: edge reduce
// g[b,i,h] = sum_j a_ij * gelu(pi[b,i,h] + pj[b,j,h] + a_ij*w1e[h] + b1[h])  (bf16 out)
__global__ void msgreduce_kernel(const float* __restrict__ pi, const float* __restrict__ pj,
                                 const float* __restrict__ adj, const float* __restrict__ Wmsg1,
                                 const float* __restrict__ bmsg1,
                                 __bf16* __restrict__ g, float* __restrict__ rowsum) {
  __shared__ float arow[NN];
  int bi = blockIdx.x;               // b*256 + i
  int b  = bi >> 8, i = bi & 255;
  int t  = threadIdx.x;              // h-dim, 0..127
  for (int j = t; j < NN; j += 128) {
    float a = adj[(size_t)bi * NN + j];
    arow[j] = (j == i) ? 0.0f : a;
  }
  __syncthreads();
  float pit = pi[(size_t)bi * ND + t];
  float w1  = Wmsg1[2 * ND * ND + t];   // row 2D of W_msg1
  float b1  = bmsg1[t];
  const float* pjb = pj + (size_t)b * NN * ND;
  float acc = 0.0f;
  for (int j = 0; j < NN; ++j) {
    float a = arow[j];
    float x = pit + pjb[(size_t)j * ND + t] + a * w1 + b1;
    acc += a * geluf(x);
  }
  g[(size_t)bi * ND + t] = f2bf(acc);
  if (t == 0) {
    float s = 0.0f;
    for (int j = 0; j < NN; ++j) s += arow[j];
    rowsum[bi] = s;
  }
}

// ---------------------------------------------------------------- E2: GRU gates
__global__ void gru_kernel(const float* __restrict__ gi, const float* __restrict__ gh,
                           float* __restrict__ h, __bf16* __restrict__ h_bf) {
  int idx = blockIdx.x * blockDim.x + threadIdx.x;
  if (idx >= NB * NN * ND) return;
  int m = idx >> 7, d = idx & 127;
  const float* gim = gi + (size_t)m * 3 * ND;
  const float* ghm = gh + (size_t)m * 3 * ND;
  float r = sigmf(gim[d] + ghm[d]);
  float z = sigmf(gim[ND + d] + ghm[ND + d]);
  float n = tanhf(gim[2 * ND + d] + r * ghm[2 * ND + d]);
  float ho = h[idx];
  float hn = (1.0f - z) * n + z * ho;
  h[idx] = hn; h_bf[idx] = f2bf(hn);
}

// ---------------------------------------------------------------- E3: pair LN+gelu -> bf16
__global__ void pairln_kernel(const float* __restrict__ A1, const float* __restrict__ A2,
                              const int* __restrict__ iu, const int* __restrict__ ju,
                              const float* __restrict__ lng, const float* __restrict__ lnb,
                              __bf16* __restrict__ peact, int base) {
  __shared__ float red[256];
  __shared__ float red2[256];
  int r = base + blockIdx.x;
  int b = r / NPAIR, p = r % NPAIR;
  int i = iu[p], j = ju[p];
  int c = threadIdx.x;
  float x = A1[((size_t)(b * NN + i)) * 256 + c] + A2[((size_t)(b * NN + j)) * 256 + c];
  red[c] = x; red2[c] = x * x;
  __syncthreads();
  for (int s = 128; s > 0; s >>= 1) {
    if (c < s) { red[c] += red[c + s]; red2[c] += red2[c + s]; }
    __syncthreads();
  }
  float mu  = red[0] * (1.0f / 256.0f);
  float var = red2[0] * (1.0f / 256.0f) - mu * mu;
  float pe  = (x - mu) * rsqrtf(var + 1e-5f) * lng[c] + lnb[c];
  peact[(size_t)blockIdx.x * 256 + c] = f2bf(geluf(pe));
}

// ---------------------------------------------------------------- E4: bilinear dots
// comb[.,128+k] = va . Tv[b,j,k,:] + bv[k] ; comb[.,192+k] = ca . Tc[b,j,k,:] + bc[k]
__global__ void pairvc_kernel(const float* __restrict__ bond, const float* __restrict__ chg,
                              const float* __restrict__ Tv, const float* __restrict__ Tc,
                              const int* __restrict__ iu, const int* __restrict__ ju,
                              const float* __restrict__ bv, const float* __restrict__ bc,
                              __bf16* __restrict__ comb, int base) {
  __shared__ float va[32];
  __shared__ float ca[16];
  int r = base + blockIdx.x;
  int b = r / NPAIR, p = r % NPAIR;
  int i = iu[p], j = ju[p];
  int t = threadIdx.x;
  if (t < 32)       va[t]      = bond[(((size_t)(b * NN + i)) * 4 + 0) * 32 + t];
  else if (t < 48)  ca[t - 32] = chg[((size_t)(b * NN + i)) * 16 + (t - 32)];
  __syncthreads();
  __bf16* crow = comb + (size_t)blockIdx.x * 224;
  if (t < 64) {
    const float* tr = Tv + ((size_t)(b * NN + j)) * 2048 + t * 32;
    float acc = 0.0f;
#pragma unroll
    for (int mm = 0; mm < 32; ++mm) acc += va[mm] * tr[mm];
    crow[128 + t] = f2bf(acc + bv[t]);
  } else if (t < 96) {
    int k = t - 64;
    const float* tr = Tc + ((size_t)(b * NN + j)) * 512 + k * 16;
    float acc = 0.0f;
#pragma unroll
    for (int mm = 0; mm < 16; ++mm) acc += ca[mm] * tr[mm];
    crow[192 + k] = f2bf(acc + bc[k]);
  }
}

// ---------------------------------------------------------------- E5: heads
__global__ void head_kernel(const float* __restrict__ hid,
                            const float* __restrict__ Ws, const float* __restrict__ bs,
                            const float* __restrict__ Wt, const float* __restrict__ bt,
                            float* __restrict__ outS, float* __restrict__ outT,
                            int base, int rows) {
  int lr = blockIdx.x * blockDim.x + threadIdx.x;
  if (lr >= rows) return;
  int r = base + lr;
  const float* hr = hid + (size_t)lr * 32;
  float s = bs[0];
  float t0 = bt[0], t1 = bt[1], t2 = bt[2], t3 = bt[3];
#pragma unroll
  for (int m = 0; m < 32; ++m) {
    float h = hr[m];
    s  += h * Ws[m];
    t0 += h * Wt[m * 4 + 0]; t1 += h * Wt[m * 4 + 1];
    t2 += h * Wt[m * 4 + 2]; t3 += h * Wt[m * 4 + 3];
  }
  outS[r] = sigmf(s);
  outT[(size_t)r * 4 + 0] = t0; outT[(size_t)r * 4 + 1] = t1;
  outT[(size_t)r * 4 + 2] = t2; outT[(size_t)r * 4 + 3] = t3;
}

// ---------------------------------------------------------------- launch
extern "C" void kernel_launch(void* const* d_in, const int* in_sizes, int n_in,
                              void* d_out, int out_size, void* d_ws, size_t ws_size,
                              hipStream_t stream) {
  const float* flat  = (const float*)d_in[0];
  const float* bond  = (const float*)d_in[1];
  const float* chg   = (const float*)d_in[2];
  const float* adj   = (const float*)d_in[3];
  const float* Wmsg1 = (const float*)d_in[4];
  const float* bmsg1 = (const float*)d_in[5];
  const float* Wmsg2 = (const float*)d_in[6];
  const float* bmsg2 = (const float*)d_in[7];
  const float* W_ih  = (const float*)d_in[8];
  const float* W_hh  = (const float*)d_in[9];
  const float* b_ih  = (const float*)d_in[10];
  const float* b_hh  = (const float*)d_in[11];
  const float* Wp1   = (const float*)d_in[12];
  const float* bp1   = (const float*)d_in[13];
  const float* ln_g  = (const float*)d_in[14];
  const float* ln_b  = (const float*)d_in[15];
  const float* Wp2   = (const float*)d_in[16];
  const float* bp2   = (const float*)d_in[17];
  const float* Wv    = (const float*)d_in[18];
  const float* bv    = (const float*)d_in[19];
  const float* Wc    = (const float*)d_in[20];
  const float* bc    = (const float*)d_in[21];
  const float* Wh1   = (const float*)d_in[22];
  const float* bh1   = (const float*)d_in[23];
  const float* Wh2   = (const float*)d_in[24];
  const float* bh2   = (const float*)d_in[25];
  const float* Ws    = (const float*)d_in[26];
  const float* bs    = (const float*)d_in[27];
  const float* Wt    = (const float*)d_in[28];
  const float* bt    = (const float*)d_in[29];

  // workspace carve-up (256B aligned)
  char* wp = (char*)d_ws;
  auto alloc = [&](size_t bytes) { char* r = wp; wp += (bytes + 255) & ~(size_t)255; return r; };
  const int M = NB * NN;                                           // 1024 node rows
  float*  h       = (float*)alloc((size_t)M * ND * 4);
  __bf16* h_bf    = (__bf16*)alloc((size_t)M * ND * 2);
  float*  pi      = (float*)alloc((size_t)M * ND * 4);
  float*  pj      = (float*)alloc((size_t)M * ND * 4);
  __bf16* g       = (__bf16*)alloc((size_t)M * ND * 2);
  __bf16* msgs    = (__bf16*)alloc((size_t)M * ND * 2);
  float*  rowsum  = (float*)alloc((size_t)M * 4);
  float*  gi      = (float*)alloc((size_t)M * 3 * ND * 4);
  float*  gh      = (float*)alloc((size_t)M * 3 * ND * 4);
  float*  A1      = (float*)alloc((size_t)M * 256 * 4);
  float*  A2      = (float*)alloc((size_t)M * 256 * 4);
  float*  Tv      = (float*)alloc((size_t)M * 2048 * 4);           // 8 MB
  float*  Tc      = (float*)alloc((size_t)M * 512 * 4);            // 2 MB
  __bf16* comb    = (__bf16*)alloc((size_t)CHUNK * 224 * 2);       // 7.3 MB
  __bf16* hid1    = (__bf16*)alloc((size_t)CHUNK * 64 * 2);
  float*  hid     = (float*)alloc((size_t)CHUNK * 32 * 4);
  __bf16* peact   = (__bf16*)alloc((size_t)CHUNK * 256 * 2);       // 8 MB
  int*    iu      = (int*)alloc((size_t)NPAIR * 4);
  int*    ju      = (int*)alloc((size_t)NPAIR * 4);
  __bf16* abond   = (__bf16*)alloc((size_t)M * 32 * 2);
  __bf16* achg    = (__bf16*)alloc((size_t)M * 32 * 2);
  // pre-transposed bf16 weights BT[n][Kpad]
  __bf16* BT1i  = (__bf16*)alloc((size_t)128 * 128 * 2);
  __bf16* BT1j  = (__bf16*)alloc((size_t)128 * 128 * 2);
  __bf16* BT2   = (__bf16*)alloc((size_t)128 * 128 * 2);
  __bf16* BTih  = (__bf16*)alloc((size_t)384 * 128 * 2);
  __bf16* BThh  = (__bf16*)alloc((size_t)384 * 128 * 2);
  __bf16* BTp1a = (__bf16*)alloc((size_t)256 * 128 * 2);
  __bf16* BTp1b = (__bf16*)alloc((size_t)256 * 128 * 2);
  __bf16* BTp2  = (__bf16*)alloc((size_t)128 * 256 * 2);
  __bf16* BTv   = (__bf16*)alloc((size_t)2048 * 32 * 2);
  __bf16* BTc   = (__bf16*)alloc((size_t)512 * 32 * 2);
  __bf16* BTh1  = (__bf16*)alloc((size_t)64 * 224 * 2);
  __bf16* BTh2  = (__bf16*)alloc((size_t)32 * 64 * 2);

  auto makeBT = [&](const float* W, int ldw, bool trans, int K, int Kpad, int N, __bf16* BT) {
    int tot = N * Kpad;
    makeBT_kernel<<<dim3((tot + 255) / 256), dim3(256), 0, stream>>>(W, ldw, (int)trans, K, Kpad, N, BT);
  };
  makeBT(Wmsg1,            128, false, 128, 128, 128, BT1i);
  makeBT(Wmsg1 + 128*128,  128, false, 128, 128, 128, BT1j);
  makeBT(Wmsg2,            128, false, 128, 128, 128, BT2);
  makeBT(W_ih,             128, true,  128, 128, 384, BTih);   // gi = msgs @ W_ih^T
  makeBT(W_hh,             128, true,  128, 128, 384, BThh);
  makeBT(Wp1,              256, false, 128, 128, 256, BTp1a);  // Wp1 rows [0:128)
  makeBT(Wp1 + 128*256,    256, false, 128, 128, 256, BTp1b);  // Wp1 rows [128:256)
  makeBT(Wp2,              128, false, 256, 256, 128, BTp2);
  makeBT(Wv,                32, true,   32,  32, 2048, BTv);   // BT[(k,i)][m]=Wv[k,i,m]
  makeBT(Wc,                16, true,   16,  32, 512, BTc);    // K padded 16->32
  makeBT(Wh1,               64, false, 224, 224, 64, BTh1);
  makeBT(Wh2,               32, false,  64,  64, 32, BTh2);

  inith_kernel<<<dim3((M * ND + 255) / 256), dim3(256), 0, stream>>>(flat, h, h_bf);
  initvc_kernel<<<dim3((M * 32 + 255) / 256), dim3(256), 0, stream>>>(bond, chg, abond, achg);
  pairidx_kernel<<<dim3((NPAIR + 255) / 256), dim3(256), 0, stream>>>(iu, ju);

  for (int it = 0; it < NITERS; ++it) {
    // pi = h @ W1i ; pj = h @ W1j
    gemm(stream, M, ND, 128, h_bf, ND, BT1i, 128, nullptr, nullptr, pi, nullptr, ND, false);
    gemm(stream, M, ND, 128, h_bf, ND, BT1j, 128, nullptr, nullptr, pj, nullptr, ND, false);
    // g = sum_j a_ij * gelu(...) ; rowsum = sum_j a_ij
    msgreduce_kernel<<<dim3(M), dim3(ND), 0, stream>>>(pi, pj, adj, Wmsg1, bmsg1, g, rowsum);
    // messages = g @ W_msg2 + rowsum[m]*b_msg2[n]   (bf16 out, feeds next GEMM)
    gemm(stream, M, ND, 128, g, ND, BT2, 128, bmsg2, rowsum, nullptr, msgs, ND, false);
    // gi = messages @ W_ih^T + b_ih ; gh = h @ W_hh^T + b_hh
    gemm(stream, M, 3 * ND, 128, msgs, ND, BTih, 128, b_ih, nullptr, gi, nullptr, 3 * ND, false);
    gemm(stream, M, 3 * ND, 128, h_bf, ND, BThh, 128, b_hh, nullptr, gh, nullptr, 3 * ND, false);
    gru_kernel<<<dim3((M * ND + 255) / 256), dim3(256), 0, stream>>>(gi, gh, h, h_bf);
  }

  // pair head precompute: A1 = h@Wp1_top + bp1 ; A2 = h@Wp1_bot
  gemm(stream, M, 256, 128, h_bf, ND, BTp1a, 128, bp1, nullptr, A1, nullptr, 256, false);
  gemm(stream, M, 256, 128, h_bf, ND, BTp1b, 128, nullptr, nullptr, A2, nullptr, 256, false);
  // Tv[b,n,(k,i)] = sum_m Wv[k,i,m]*bond[b,n,0,m] ; Tc analog (K padded to 32)
  gemm(stream, M, 2048, 32, abond, 32, BTv, 32, nullptr, nullptr, Tv, nullptr, 2048, false);
  gemm(stream, M, 512, 32, achg, 32, BTc, 32, nullptr, nullptr, Tc, nullptr, 512, false);

  float* outS = (float*)d_out;            // (B,P) strengths
  float* outT = outS + TOTROWS;           // (B,P,4) type logits

  for (int base = 0; base < TOTROWS; base += CHUNK) {
    int rows = TOTROWS - base; if (rows > CHUNK) rows = CHUNK;   // 16384 / 15872, both %16==0
    pairln_kernel<<<dim3(rows), dim3(256), 0, stream>>>(A1, A2, iu, ju, ln_g, ln_b, peact, base);
    // pe = gelu(LN(pair@Wp1)) @ Wp2 + bp2 -> comb[:,0:128]  (bf16)
    gemm(stream, rows, 128, 256, peact, 256, BTp2, 256, bp2, nullptr, nullptr, comb, 224, false);
    pairvc_kernel<<<dim3(rows), dim3(128), 0, stream>>>(bond, chg, Tv, Tc, iu, ju, bv, bc, comb, base);
    // hid1 = gelu(comb @ Wh1 + bh1)  (bf16) ; hid = hid1 @ Wh2 + bh2 (f32)
    gemm(stream, rows, 64, 224, comb, 224, BTh1, 224, bh1, nullptr, nullptr, hid1, 64, true);
    gemm(stream, rows, 32, 64,  hid1, 64,  BTh2, 64,  bh2, nullptr, hid, nullptr, 32, false);
    head_kernel<<<dim3((rows + 255) / 256), dim3(256), 0, stream>>>(hid, Ws, bs, Wt, bt, outS, outT, base, rows);
  }
}
